// GINEEncoder_43559558316462
// MI455X (gfx1250) — matrix operand order repaired and checked by reading, not compiled, not run on hardware
//
#include <hip/hip_runtime.h>

// ---------------------------------------------------------------------------
// GINE encoder forward for MI455X (gfx1250, wave32, WMMA).
// Per layer:
//   0) zero agg + BN stats
//   1) edge kernel : e = f16WMMA(edge_attr @ We) + be ; msg = relu(h[src]+e);
//                    atomic scatter-add into agg[dst]        (WMMA 16x16x32)
//   2) node kernel : z = relu((h+agg)@W1 + b1)@W2 + b2 ; BN partial sums
//   3) BN stats finalize (mu, rsqrt(var+eps))
//   4) BN apply + relu + 0.1*residual -> next h (last layer -> d_out)
// One-time prep: weights -> f16 B-layout swizzle; edge_attr -> f16 A-layout
// swizzle; int64 edge_index -> int32 element offsets. WMMAs use a shared
// zero C (bias folded into the epilogue add) to keep register pressure low,
// 32-bit saddr+scale_offset addressing, and batched gathers for MLP.
// ---------------------------------------------------------------------------

typedef _Float16 v16h __attribute__((ext_vector_type(16)));
typedef float    v8f  __attribute__((ext_vector_type(8)));

#define N_NODES 50000
#define N_EDGES 800000
#define HID     128
#define EDIM    32
#define NL      3
#define BN_EPS  1e-5f
#define RES_SCALE 0.1f

// ---------------------------------------------------------------------------
// Weight pre-swizzle: f32 row-major [K][128] -> f16 B-layout tiles.
// W1/W2 dest: ((t*4+kc)*32 + lane)*16 + e, K = kc*32+(e>>3)*16+(lane>>4)*8+(e&7),
//             col = t*16+(lane&15).  We (K=32): (t*32+lane)*16+e.
// ---------------------------------------------------------------------------
__global__ __launch_bounds__(256) void gine_prep_weights(
    const float* __restrict__ We, const float* __restrict__ W1,
    const float* __restrict__ W2,
    _Float16* __restrict__ WeSw, _Float16* __restrict__ W1Sw,
    _Float16* __restrict__ W2Sw)
{
    const int i = blockIdx.x * blockDim.x + threadIdx.x;
    if (i < NL * 4096) {
        const int l = i / 4096, r = i % 4096;
        const int e = r & 15, ln = (r >> 4) & 31, t = r >> 9;
        const int kin = ((e >> 3) * 16) + ((ln >> 4) * 8) + (e & 7);
        const int col = t * 16 + (ln & 15);
        WeSw[i] = (_Float16)We[l * EDIM * HID + kin * HID + col];
    }
    if (i < NL * 16384) {
        const int l = i / 16384, r = i % 16384;
        const int e = r & 15, ln = (r >> 4) & 31, kc = (r >> 9) & 3, t = r >> 11;
        const int kin = kc * 32 + ((e >> 3) * 16) + ((ln >> 4) * 8) + (e & 7);
        const int col = t * 16 + (ln & 15);
        W1Sw[i] = (_Float16)W1[l * HID * HID + kin * HID + col];
        W2Sw[i] = (_Float16)W2[l * HID * HID + kin * HID + col];
    }
}

// ---------------------------------------------------------------------------
// Edge prep: int64 indices -> int32 element offsets (id*128); edge_attr ->
// f16 A-layout tiles: dest (tile*32+lane)*16+e holds
//   row = tile*16+(lane&15), K = (e<8 ? (lane>>4)*8+e : 16+(lane>>4)*8+e-8)
// ---------------------------------------------------------------------------
__global__ __launch_bounds__(256) void gine_prep_edges(
    const long long* __restrict__ ei,
    const float* __restrict__ edge_attr,
    int* __restrict__ srcOff, int* __restrict__ dstOff,
    _Float16* __restrict__ eaSw)
{
    const int i = blockIdx.x * blockDim.x + threadIdx.x;
    if (i < N_EDGES) {
        srcOff[i] = (int)ei[i] * HID;
        dstOff[i] = (int)ei[N_EDGES + i] * HID;
    }
    if (i < N_EDGES * EDIM) {
        const int e = i & 15, ln = (i >> 4) & 31, tile = i >> 9;
        const int row = tile * 16 + (ln & 15);
        const int kb = (ln >> 4) * 8;
        const int K = (e < 8) ? (kb + e) : (16 + kb + (e - 8));
        eaSw[i] = (_Float16)edge_attr[(size_t)row * EDIM + K];
    }
}

__global__ __launch_bounds__(256) void gine_zero(float* __restrict__ p, int count)
{
    const int i = blockIdx.x * blockDim.x + threadIdx.x;
    if (i < count) p[i] = 0.0f;
}

// ---------------------------------------------------------------------------
// Edge kernel: one wave processes several 16-edge x 128 tiles (grid-stride)
// so the B (We) registers / biases are loaded once per wave. Per tile:
// A = one contiguous v16h, 8 WMMAs with shared zero C, then batched
// gather -> (+e +bias, relu) -> atomic scatter epilogue using 32-bit offsets
// (saddr + scale_offset + imm t*64B). Wave-uniform loop -> EXEC all-1s.
// ---------------------------------------------------------------------------
__global__ __launch_bounds__(256) void gine_edge_kernel(
    const float* __restrict__ h,
    const _Float16* __restrict__ eaSw,     // [E/16][32][16] A-swizzled
    const int* __restrict__ srcOff,        // [E] element offsets (id*128)
    const int* __restrict__ dstOff,        // [E]
    const _Float16* __restrict__ WeSw,     // [8][32][16] B-swizzled
    const float* __restrict__ be_l,        // [128]
    float* __restrict__ agg)               // [N_NODES][128]
{
    const int lane = threadIdx.x & 31;
    const int wave = threadIdx.x >> 5;
    const int n    = lane & 15;
    const int hi   = lane >> 4;

    v16h b[8];
#pragma unroll
    for (int t = 0; t < 8; ++t)
        b[t] = *(const v16h*)(WeSw + (t * 32 + lane) * 16);

    const float* pbe = be_l + n;
    float bias[8];
#pragma unroll
    for (int t = 0; t < 8; ++t) bias[t] = pbe[t * 16];

    const v8f czero = {};

    const int numTiles = N_EDGES / 16;     // 50000
    const int nwaves   = gridDim.x * 8;
    for (int tile = blockIdx.x * 8 + wave; tile < numTiles; tile += nwaves) {
        const int ebase = tile * 16;
        const v16h a = *(const v16h*)(eaSw + (tile * 32 + lane) * 16);

        // The 8 C-rows this lane owns: M = r + hi*8
        const int* sp = srcOff + ebase + hi * 8;
        const int* dp = dstOff + ebase + hi * 8;
        int so[8], dofs[8];
#pragma unroll
        for (int r = 0; r < 8; ++r) {
            so[r]   = sp[r] + n;           // 32-bit element offsets (col n)
            dofs[r] = dp[r] + n;
        }

#pragma unroll
        for (int t = 0; t < 8; ++t) {
            v8f c = __builtin_amdgcn_wmma_f32_16x16x32_f16(
                        false, a, false, b[t], (short)0, czero, false, false);
            // batched gathers: one clause, one wait
            float g[8];
#pragma unroll
            for (int r = 0; r < 8; ++r) g[r] = h[so[r] + t * 16];
#pragma unroll
            for (int r = 0; r < 8; ++r) {
                float v = g[r] + c[r] + bias[t];
                v = v > 0.0f ? v : 0.0f;
                atomicAdd(&agg[dofs[r] + t * 16], v);
            }
        }
    }
}

// ---------------------------------------------------------------------------
// Node kernel: one wave = 16 nodes x 128. Two chained WMMA GEMMs (K=128 as
// 4x K=32, zero-init C, bias in epilogue). z1 bounces through per-wave LDS
// to convert C-layout -> A-layout (same-wave DS ops are in-order, no barrier
// needed). Epilogues use base pointers + immediate offsets; BN partial sums.
// ---------------------------------------------------------------------------
__global__ __launch_bounds__(256) void gine_node_kernel(
    const float* __restrict__ h,
    const float* __restrict__ agg,
    const _Float16* __restrict__ W1Sw,     // [8][4][32][16] B-swizzled
    const float* __restrict__ b1_l,
    const _Float16* __restrict__ W2Sw,
    const float* __restrict__ b2_l,
    float* __restrict__ z2,                // [N_NODES][128]
    float* __restrict__ sumBuf,            // [128]
    float* __restrict__ sqBuf)             // [128]
{
    __shared__ _Float16 sZ1[8 * 16 * HID]; // 4KB per wave, 32KB total
    const int lane = threadIdx.x & 31;
    const int wave = threadIdx.x >> 5;
    const int n    = lane & 15;
    const int hi   = lane >> 4;
    const int kb   = hi * 8;
    _Float16* z1 = sZ1 + wave * (16 * HID);

    const float* pb1 = b1_l + n;
    const float* pb2 = b2_l + n;
    float* psum = sumBuf + n;
    float* psq  = sqBuf  + n;

    // LDS bases for C-layout store (rows hi*8..hi*8+7) and A-layout reload
    _Float16*       zst = z1 + (hi * 8) * HID + n;
    const _Float16* zld = z1 + n * HID + kb;

    const v8f czero = {};

    const int numTiles = N_NODES / 16;     // 3125
    for (int tile = blockIdx.x * 8 + wave; tile < numTiles;
         tile += gridDim.x * 8) {
        const int nbase = tile * 16;

        // A = f16(h + agg), row = nbase + n
        const float* hp = h   + (size_t)(nbase + n) * HID + kb;
        const float* ap = agg + (size_t)(nbase + n) * HID + kb;
        v16h a[4];
#pragma unroll
        for (int c = 0; c < 4; ++c) {
#pragma unroll
            for (int e = 0; e < 8; ++e)
                a[c][e]     = (_Float16)(hp[c * 32 + e] + ap[c * 32 + e]);
#pragma unroll
            for (int e = 0; e < 8; ++e)
                a[c][8 + e] = (_Float16)(hp[c * 32 + 16 + e] + ap[c * 32 + 16 + e]);
        }

        // GEMM1: z1 = relu(A @ W1 + b1)  -> LDS (f16, row-major 16x128)
#pragma unroll
        for (int t = 0; t < 8; ++t) {
            v8f cc = czero;
#pragma unroll
            for (int kc = 0; kc < 4; ++kc) {
                v16h bb = *(const v16h*)(W1Sw + ((t * 4 + kc) * 32 + lane) * 16);
                cc = __builtin_amdgcn_wmma_f32_16x16x32_f16(
                         false, a[kc], false, bb, (short)0, cc, false, false);
            }
            const float bias = pb1[t * 16];
#pragma unroll
            for (int r = 0; r < 8; ++r) {
                float v = cc[r] + bias;
                v = v > 0.0f ? v : 0.0f;
                zst[r * HID + t * 16] = (_Float16)v;   // imm ds offset
            }
        }

        // Reload z1 in A-layout (contiguous 8-half runs -> ds_load_b128)
        v16h a2[4];
#pragma unroll
        for (int c = 0; c < 4; ++c) {
#pragma unroll
            for (int e = 0; e < 8; ++e) a2[c][e]     = zld[c * 32 + e];
#pragma unroll
            for (int e = 0; e < 8; ++e) a2[c][8 + e] = zld[c * 32 + 16 + e];
        }

        // GEMM2: z2 = z1 @ W2 + b2 ; BN partial sums
        float* pz = z2 + (size_t)(nbase + hi * 8) * HID + n;
#pragma unroll
        for (int t = 0; t < 8; ++t) {
            v8f cc = czero;
#pragma unroll
            for (int kc = 0; kc < 4; ++kc) {
                v16h bb = *(const v16h*)(W2Sw + ((t * 4 + kc) * 32 + lane) * 16);
                cc = __builtin_amdgcn_wmma_f32_16x16x32_f16(
                         false, a2[kc], false, bb, (short)0, cc, false, false);
            }
            const float bias = pb2[t * 16];
            float ls = 0.0f, lq = 0.0f;
#pragma unroll
            for (int r = 0; r < 8; ++r) {
                const float v = cc[r] + bias;
                pz[r * HID + t * 16] = v;              // imm offset
                ls += v; lq += v * v;
            }
            atomicAdd(psum + t * 16, ls);              // imm offset
            atomicAdd(psq  + t * 16, lq);              // imm offset
        }
    }
}

__global__ void gine_bn_stats(const float* __restrict__ sumBuf,
                              const float* __restrict__ sqBuf,
                              float* __restrict__ muB,
                              float* __restrict__ invB)
{
    const int c = threadIdx.x;                 // 128 threads
    const float m = sumBuf[c] * (1.0f / (float)N_NODES);
    const float v = sqBuf[c] * (1.0f / (float)N_NODES) - m * m;
    muB[c]  = m;
    invB[c] = rsqrtf(v + BN_EPS);
}

__global__ __launch_bounds__(256) void gine_bn_apply(
    const float* __restrict__ z2,
    const float* __restrict__ muB, const float* __restrict__ invB,
    const float* __restrict__ gamma_l, const float* __restrict__ beta_l,
    const float* __restrict__ h_res, float* __restrict__ out)
{
    const int i = blockIdx.x * blockDim.x + threadIdx.x;   // one float4
    const int total = N_NODES * HID / 4;
    if (i >= total) return;
    const int c4 = (i & (HID / 4 - 1)) * 4;
    const float4 z = ((const float4*)z2)[i];
    const float4 r = ((const float4*)h_res)[i];
    float4 o;
    float v;
    v = (z.x - muB[c4 + 0]) * invB[c4 + 0] * gamma_l[c4 + 0] + beta_l[c4 + 0];
    o.x = (v > 0.0f ? v : 0.0f) + RES_SCALE * r.x;
    v = (z.y - muB[c4 + 1]) * invB[c4 + 1] * gamma_l[c4 + 1] + beta_l[c4 + 1];
    o.y = (v > 0.0f ? v : 0.0f) + RES_SCALE * r.y;
    v = (z.z - muB[c4 + 2]) * invB[c4 + 2] * gamma_l[c4 + 2] + beta_l[c4 + 2];
    o.z = (v > 0.0f ? v : 0.0f) + RES_SCALE * r.z;
    v = (z.w - muB[c4 + 3]) * invB[c4 + 3] * gamma_l[c4 + 3] + beta_l[c4 + 3];
    o.w = (v > 0.0f ? v : 0.0f) + RES_SCALE * r.w;
    ((float4*)out)[i] = o;
}

// ---------------------------------------------------------------------------
extern "C" void kernel_launch(void* const* d_in, const int* in_sizes, int n_in,
                              void* d_out, int out_size, void* d_ws, size_t ws_size,
                              hipStream_t stream)
{
    (void)in_sizes; (void)n_in; (void)out_size; (void)ws_size;
    const float*      x         = (const float*)d_in[0];
    const long long*  ei        = (const long long*)d_in[1];   // int64 [2][E]
    const float*      edge_attr = (const float*)d_in[2];
    const float*      We        = (const float*)d_in[3];
    const float*      be        = (const float*)d_in[4];
    const float*      W1        = (const float*)d_in[5];
    const float*      b1        = (const float*)d_in[6];
    const float*      W2        = (const float*)d_in[7];
    const float*      b2        = (const float*)d_in[8];
    const float*      gamma     = (const float*)d_in[9];
    const float*      beta      = (const float*)d_in[10];
    float*            out       = (float*)d_out;

    // ws layout (floats): hA | hB | z2 | agg | stats(512) | f16 wts | idx | eaSw
    float* ws    = (float*)d_ws;
    const size_t NH = (size_t)N_NODES * HID;
    float* hA    = ws;
    float* hB    = hA + NH;
    float* z2    = hB + NH;
    float* agg   = z2 + NH;
    float* stats = agg + NH;
    float* sumBuf = stats;
    float* sqBuf  = stats + 128;
    float* muB    = stats + 256;
    float* invB   = stats + 384;
    _Float16* WeSw = (_Float16*)(stats + 512);
    _Float16* W1Sw = WeSw + NL * 4096;
    _Float16* W2Sw = W1Sw + NL * 16384;
    int* srcOff    = (int*)(W2Sw + NL * 16384);
    int* dstOff    = srcOff + N_EDGES;
    _Float16* eaSw = (_Float16*)(dstOff + N_EDGES);  // [E*32] halfs

    gine_prep_weights<<<(NL * 16384 + 255) / 256, 256, 0, stream>>>(
        We, W1, W2, WeSw, W1Sw, W2Sw);
    gine_prep_edges<<<(N_EDGES * EDIM + 255) / 256, 256, 0, stream>>>(
        ei, edge_attr, srcOff, dstOff, eaSw);

    const float* hin = x;
    for (int l = 0; l < NL; ++l) {
        // zero agg + stats (contiguous: agg .. stats+512)
        const int zc = (int)NH + 512;
        gine_zero<<<(zc + 255) / 256, 256, 0, stream>>>(agg, zc);

        // ~4 tiles per wave: 1563 blocks * 8 waves covers 50000 tiles
        gine_edge_kernel<<<1563, 256, 0, stream>>>(
            hin, eaSw, srcOff, dstOff, WeSw + l * 4096, be + l * HID, agg);

        gine_node_kernel<<<391, 256, 0, stream>>>(
            hin, agg, W1Sw + l * 16384, b1 + l * HID,
            W2Sw + l * 16384, b2 + l * HID, z2, sumBuf, sqBuf);

        gine_bn_stats<<<1, 128, 0, stream>>>(sumBuf, sqBuf, muB, invB);

        float* hout = (l == NL - 1) ? out : ((l == 0) ? hA : hB);
        gine_bn_apply<<<((int)NH / 4 + 255) / 256, 256, 0, stream>>>(
            z2, muB, invB, gamma + l * HID, beta + l * HID, hin, hout);
        hin = hout;
    }
}
